// RocketLatentODEPINN_53601191854788
// MI455X (gfx1250) — matrix operand order repaired and checked by reading, not compiled, not run on hardware
//
#include <hip/hip_runtime.h>
#include <hip/hip_bf16.h>

typedef __attribute__((ext_vector_type(16))) _Float16 v16h;
typedef __attribute__((ext_vector_type(8)))  float    v8f;

// ---------------------------------------------------------------------------
// WMMA helpers (CDNA5 wave32, V_WMMA_F32_16X16X32_F16)
// ---------------------------------------------------------------------------
__device__ __forceinline__ v8f wmma32(v16h a, v16h b, v8f c) {
  return __builtin_amdgcn_wmma_f32_16x16x32_f16(false, a, false, b, (short)0, c,
                                                false, false);
}

// A-fragment (16x32 f16) from LDS, row-major, leading dim = lda (halves).
// ISA layout: lanes 0-15 -> M=lane, halves[0..7]=K[k0+klo..], halves[8..15]=K[k0+16+klo..]
// where klo = 8 * (lane>=16).
__device__ __forceinline__ v16h load_a_frag(const _Float16* A, int lda, int k0,
                                            int l16, int half_) {
  v16h a;
  const _Float16* p = A + l16 * lda + k0 + half_ * 8;
#pragma unroll
  for (int i = 0; i < 8; ++i) { a[i] = p[i]; a[8 + i] = p[16 + i]; }
  return a;
}

// B-fragment (32x16 f16) built straight from a global fp32 row-major [K x N]
// weight matrix, with zero-padding past K/N. Lane = column n; halves cover
// halves[0..7]=K[k0+8*half..], halves[8..15]=K[k0+16+8*half..].  (done once per kernel; fragments persist in VGPRs)
__device__ __forceinline__ v16h load_b_frag_global(const float* __restrict__ W,
                                                   int K, int N, int k0, int n,
                                                   int half_) {
  v16h b;
  const int kk = k0 + 8 * half_;
#pragma unroll
  for (int i = 0; i < 16; ++i) {
    const int k = kk + (i < 8 ? i : 8 + i);
    float w = (k < K && n < N) ? W[k * N + n] : 0.0f;
    b[i] = (_Float16)w;
  }
  return b;
}

// LayerNorm + tanh over [16 rows x 128], 16 threads/row (8 cols each).
__device__ __forceinline__ void ln_tanh_16x128(float (*H)[128], _Float16 (*Act)[128],
                                               const float* g, const float* be,
                                               int tid) {
  const int r  = tid >> 4;
  const int c0 = (tid & 15) * 8;
  float s = 0.f, ss = 0.f;
#pragma unroll
  for (int i = 0; i < 8; ++i) { float v = H[r][c0 + i]; s += v; ss += v * v; }
#pragma unroll
  for (int m = 1; m < 16; m <<= 1) {
    s  += __shfl_xor(s,  m, 32);
    ss += __shfl_xor(ss, m, 32);
  }
  const float mu   = s * (1.0f / 128.0f);
  const float rstd = rsqrtf(ss * (1.0f / 128.0f) - mu * mu + 1e-5f);
#pragma unroll
  for (int i = 0; i < 8; ++i) {
    float y = (H[r][c0 + i] - mu) * rstd * g[c0 + i] + be[c0 + i];
    Act[r][c0 + i] = (_Float16)tanhf(y);
  }
}

// LayerNorm + tanh over [64 rows x 128], 4 threads/row (32 cols each).
__device__ __forceinline__ void ln_tanh_64x128(float (*H)[128], _Float16 (*Act)[128],
                                               const float* g, const float* be,
                                               int tid) {
  const int r  = tid >> 2;
  const int c0 = (tid & 3) * 32;
  float s = 0.f, ss = 0.f;
  for (int i = 0; i < 32; ++i) { float v = H[r][c0 + i]; s += v; ss += v * v; }
  s  += __shfl_xor(s, 1, 32);  ss += __shfl_xor(ss, 1, 32);
  s  += __shfl_xor(s, 2, 32);  ss += __shfl_xor(ss, 2, 32);
  const float mu   = s * (1.0f / 128.0f);
  const float rstd = rsqrtf(ss * (1.0f / 128.0f) - mu * mu + 1e-5f);
  for (int i = 0; i < 32; ++i) {
    float y = (H[r][c0 + i] - mu) * rstd * g[c0 + i] + be[c0 + i];
    Act[r][c0 + i] = (_Float16)tanhf(y);
  }
}

// ---------------------------------------------------------------------------
// Kernel 1: persistent ODE scan. One block = 16 trajectories, 8 waves.
// Dynamics weights live in VGPRs as WMMA B-fragments for all 255 steps.
// Writes z_traj [B*N, 64] f16 into workspace.
// ---------------------------------------------------------------------------
__global__ __launch_bounds__(256, 1) void rocket_scan_kernel(
    const float* __restrict__ t, const float* __restrict__ context,
    const float* __restrict__ ce_w, const float* __restrict__ ce_b,
    const float* __restrict__ z0_w, const float* __restrict__ z0_b,
    const float* __restrict__ dw0, const float* __restrict__ db0,
    const float* __restrict__ dw1, const float* __restrict__ db1,
    const float* __restrict__ dw2, const float* __restrict__ db2,
    const float* __restrict__ dw3, const float* __restrict__ db3,
    const float* __restrict__ dg0, const float* __restrict__ dbe0,
    const float* __restrict__ dg1, const float* __restrict__ dbe1,
    const float* __restrict__ dg2, const float* __restrict__ dbe2,
    _Float16* __restrict__ ztraj) {
  // x layout per row: [ z(0..63) | t(64) | sin(65..72) | cos(73..80) | ctx(81..144) | pad->160 ]
  __shared__ _Float16 sX[16][160];
  __shared__ _Float16 sAct[16][128];
  __shared__ float    sH[16][128];
  __shared__ float    sZ[16][64];
  __shared__ float    sCtx[16][64];
  __shared__ float    sCin[16][32];
  __shared__ float    sDT[16];
  __shared__ float    sB[3 * 128 + 64];
  __shared__ float    sG[3 * 128];
  __shared__ float    sBe[3 * 128];

  const int tid   = threadIdx.x;
  const int wave  = tid >> 5;
  const int lane  = tid & 31;
  const int half_ = lane >> 4;
  const int l16   = lane & 15;
  const int b0    = blockIdx.x * 16;

  for (int i = tid; i < 128; i += 256) {
    sB[i] = db0[i];  sB[128 + i] = db1[i];  sB[256 + i] = db2[i];
    sG[i] = dg0[i];  sG[128 + i] = dg1[i];  sG[256 + i] = dg2[i];
    sBe[i] = dbe0[i]; sBe[128 + i] = dbe1[i]; sBe[256 + i] = dbe2[i];
  }
  if (tid < 64) sB[384 + tid] = db3[tid];
  if (tid < 16) {
#pragma unroll
    for (int c = 145; c < 160; ++c) sX[tid][c] = (_Float16)0.0f;
  }
  for (int i = tid; i < 16 * 32; i += 256)
    sCin[i >> 5][i & 31] = context[(size_t)b0 * 32 + i];

  // Persistent weight fragments (per wave: col-tile = wave for 128-wide layers).
  v16h w0f[5], w1f[4], w2f[4], w3f[4];
  {
    const int n1 = wave * 16 + l16;
#pragma unroll
    for (int kc = 0; kc < 5; ++kc) w0f[kc] = load_b_frag_global(dw0, 145, 128, kc * 32, n1, half_);
#pragma unroll
    for (int kc = 0; kc < 4; ++kc) w1f[kc] = load_b_frag_global(dw1, 128, 128, kc * 32, n1, half_);
#pragma unroll
    for (int kc = 0; kc < 4; ++kc) w2f[kc] = load_b_frag_global(dw2, 128, 128, kc * 32, n1, half_);
    const int n3 = (wave & 3) * 16 + l16;  // layer 4 is 64 wide -> waves 0..3
#pragma unroll
    for (int kc = 0; kc < 4; ++kc) w3f[kc] = load_b_frag_global(dw3, 128, 64, kc * 32, n3, half_);
  }
  __syncthreads();

  // Context encoder: ctx_emb = tanh(context @ ce_w + ce_b)   (fp32, tiny)
  for (int idx = tid; idx < 16 * 64; idx += 256) {
    const int r = idx >> 6, c = idx & 63;
    float acc = ce_b[c];
    for (int k = 0; k < 32; ++k) acc += sCin[r][k] * ce_w[k * 64 + c];
    const float e = tanhf(acc);
    sCtx[r][c] = e;
    sX[r][81 + c] = (_Float16)e;  // static part of the dyn input
  }
  __syncthreads();
  // z0 = tanh(ctx_emb @ z0_w + z0_b)
  for (int idx = tid; idx < 16 * 64; idx += 256) {
    const int r = idx >> 6, c = idx & 63;
    float acc = z0_b[c];
    for (int k = 0; k < 64; ++k) acc += sCtx[r][k] * z0_w[k * 64 + c];
    const float z = tanhf(acc);
    sZ[r][c] = z;
    sX[r][c] = (_Float16)z;
  }
  __syncthreads();

  for (int n = 0; n < 256; ++n) {
    // Emit current latent state z_n (f16) for the decoder kernel.
    for (int idx = tid; idx < 16 * 64; idx += 256) {
      const int r = idx >> 6, c = idx & 63;
      ztraj[((size_t)(b0 + r) * 256 + n) * 64 + c] = sX[r][c];
    }
    if (n == 255) break;

    // Fourier time embedding + dt for this step.
    if (tid < 16) {
      const int r = tid;
      const float tc = t[(size_t)(b0 + r) * 256 + n];
      const float tn = t[(size_t)(b0 + r) * 256 + n + 1];
      sDT[r]    = tn - tc;
      sX[r][64] = (_Float16)tc;
      float f = 3.14159265358979323846f;
#pragma unroll
      for (int k = 0; k < 8; ++k) {
        float sv, cv;
        __sincosf(tc * f, &sv, &cv);
        sX[r][65 + k] = (_Float16)sv;
        sX[r][73 + k] = (_Float16)cv;
        f *= 2.0f;
      }
    }
    __syncthreads();

    // dyn layer 1: [16 x 160] @ [160 x 128]
    {
      v8f acc = {};
#pragma unroll
      for (int kc = 0; kc < 5; ++kc)
        acc = wmma32(load_a_frag(&sX[0][0], 160, kc * 32, l16, half_), w0f[kc], acc);
      const int col = wave * 16 + l16;
#pragma unroll
      for (int v = 0; v < 8; ++v) sH[v + 8 * half_][col] = acc[v] + sB[col];
    }
    __syncthreads();
    ln_tanh_16x128(sH, sAct, sG, sBe, tid);
    __syncthreads();

    // dyn layer 2: [16 x 128] @ [128 x 128]
    {
      v8f acc = {};
#pragma unroll
      for (int kc = 0; kc < 4; ++kc)
        acc = wmma32(load_a_frag(&sAct[0][0], 128, kc * 32, l16, half_), w1f[kc], acc);
      const int col = wave * 16 + l16;
#pragma unroll
      for (int v = 0; v < 8; ++v) sH[v + 8 * half_][col] = acc[v] + sB[128 + col];
    }
    __syncthreads();
    ln_tanh_16x128(sH, sAct, sG + 128, sBe + 128, tid);
    __syncthreads();

    // dyn layer 3
    {
      v8f acc = {};
#pragma unroll
      for (int kc = 0; kc < 4; ++kc)
        acc = wmma32(load_a_frag(&sAct[0][0], 128, kc * 32, l16, half_), w2f[kc], acc);
      const int col = wave * 16 + l16;
#pragma unroll
      for (int v = 0; v < 8; ++v) sH[v + 8 * half_][col] = acc[v] + sB[256 + col];
    }
    __syncthreads();
    ln_tanh_16x128(sH, sAct, sG + 256, sBe + 256, tid);
    __syncthreads();

    // dyn layer 4: [16 x 128] @ [128 x 64]  -> dz in sH[:, 0..63]
    if (wave < 4) {
      v8f acc = {};
#pragma unroll
      for (int kc = 0; kc < 4; ++kc)
        acc = wmma32(load_a_frag(&sAct[0][0], 128, kc * 32, l16, half_), w3f[kc], acc);
      const int col = wave * 16 + l16;
#pragma unroll
      for (int v = 0; v < 8; ++v) sH[v + 8 * half_][col] = acc[v] + sB[384 + col];
    }
    __syncthreads();

    // Euler update: z <- z + dt * dz (fp32 master copy, f16 shadow for WMMA)
    for (int idx = tid; idx < 16 * 64; idx += 256) {
      const int r = idx >> 6, c = idx & 63;
      const float z = sZ[r][c] + sDT[r] * sH[r][c];
      sZ[r][c] = z;
      sX[r][c] = (_Float16)z;
    }
    __syncthreads();
  }
}

// ---------------------------------------------------------------------------
// Kernel 2: decoder MLP on z_traj. One block = 64 rows of [B*N], 8 waves.
// Decoder weights persist in VGPRs as WMMA B-fragments.
// ---------------------------------------------------------------------------
__global__ __launch_bounds__(256, 1) void rocket_dec_kernel(
    const _Float16* __restrict__ ztraj,
    const float* __restrict__ ew0, const float* __restrict__ eb0,
    const float* __restrict__ ew1, const float* __restrict__ eb1,
    const float* __restrict__ ew2, const float* __restrict__ eb2,
    const float* __restrict__ ew3, const float* __restrict__ eb3,
    const float* __restrict__ eg0, const float* __restrict__ ebe0,
    const float* __restrict__ eg1, const float* __restrict__ ebe1,
    const float* __restrict__ eg2, const float* __restrict__ ebe2,
    float* __restrict__ out) {
  __shared__ _Float16 sZ[64][64];
  __shared__ float    sH[64][128];
  __shared__ _Float16 sAct[64][128];
  __shared__ float    sB[3 * 128 + 16];
  __shared__ float    sG[3 * 128];
  __shared__ float    sBe[3 * 128];

  const int tid   = threadIdx.x;
  const int wave  = tid >> 5;
  const int lane  = tid & 31;
  const int half_ = lane >> 4;
  const int l16   = lane & 15;
  const size_t m0 = (size_t)blockIdx.x * 64;

  for (int i = tid; i < 128; i += 256) {
    sB[i] = eb0[i];  sB[128 + i] = eb1[i];  sB[256 + i] = eb2[i];
    sG[i] = eg0[i];  sG[128 + i] = eg1[i];  sG[256 + i] = eg2[i];
    sBe[i] = ebe0[i]; sBe[128 + i] = ebe1[i]; sBe[256 + i] = ebe2[i];
  }
  if (tid < 16) sB[384 + tid] = (tid < 14) ? eb3[tid] : 0.0f;

  v16h v0f[2], v1f[4], v2f[4], v3f[4];
  {
    const int n1 = wave * 16 + l16;
#pragma unroll
    for (int kc = 0; kc < 2; ++kc) v0f[kc] = load_b_frag_global(ew0, 64, 128, kc * 32, n1, half_);
#pragma unroll
    for (int kc = 0; kc < 4; ++kc) v1f[kc] = load_b_frag_global(ew1, 128, 128, kc * 32, n1, half_);
#pragma unroll
    for (int kc = 0; kc < 4; ++kc) v2f[kc] = load_b_frag_global(ew2, 128, 128, kc * 32, n1, half_);
#pragma unroll
    for (int kc = 0; kc < 4; ++kc) v3f[kc] = load_b_frag_global(ew3, 128, 14, kc * 32, l16, half_);
  }

  // Stage 64x64 f16 z tile (vectorized 16B copies).
  {
    const uint4* src = (const uint4*)(ztraj + m0 * 64);
    uint4* dst = (uint4*)&sZ[0][0];
    for (int i = tid; i < 64 * 64 / 8; i += 256) dst[i] = src[i];
  }
  __syncthreads();

  // layer 1: [64 x 64] @ [64 x 128]
  {
    v8f acc[4] = {};
#pragma unroll
    for (int kc = 0; kc < 2; ++kc)
#pragma unroll
      for (int ms = 0; ms < 4; ++ms)
        acc[ms] = wmma32(load_a_frag(&sZ[ms * 16][0], 64, kc * 32, l16, half_), v0f[kc], acc[ms]);
    const int col = wave * 16 + l16;
#pragma unroll
    for (int ms = 0; ms < 4; ++ms)
#pragma unroll
      for (int v = 0; v < 8; ++v)
        sH[ms * 16 + v + 8 * half_][col] = acc[ms][v] + sB[col];
  }
  __syncthreads();
  ln_tanh_64x128(sH, sAct, sG, sBe, tid);
  __syncthreads();

  // layer 2: [64 x 128] @ [128 x 128]
  {
    v8f acc[4] = {};
#pragma unroll
    for (int kc = 0; kc < 4; ++kc)
#pragma unroll
      for (int ms = 0; ms < 4; ++ms)
        acc[ms] = wmma32(load_a_frag(&sAct[ms * 16][0], 128, kc * 32, l16, half_), v1f[kc], acc[ms]);
    const int col = wave * 16 + l16;
#pragma unroll
    for (int ms = 0; ms < 4; ++ms)
#pragma unroll
      for (int v = 0; v < 8; ++v)
        sH[ms * 16 + v + 8 * half_][col] = acc[ms][v] + sB[128 + col];
  }
  __syncthreads();
  ln_tanh_64x128(sH, sAct, sG + 128, sBe + 128, tid);
  __syncthreads();

  // layer 3
  {
    v8f acc[4] = {};
#pragma unroll
    for (int kc = 0; kc < 4; ++kc)
#pragma unroll
      for (int ms = 0; ms < 4; ++ms)
        acc[ms] = wmma32(load_a_frag(&sAct[ms * 16][0], 128, kc * 32, l16, half_), v2f[kc], acc[ms]);
    const int col = wave * 16 + l16;
#pragma unroll
    for (int ms = 0; ms < 4; ++ms)
#pragma unroll
      for (int v = 0; v < 8; ++v)
        sH[ms * 16 + v + 8 * half_][col] = acc[ms][v] + sB[256 + col];
  }
  __syncthreads();
  ln_tanh_64x128(sH, sAct, sG + 256, sBe + 256, tid);
  __syncthreads();

  // layer 4: [64 x 128] @ [128 x 14(pad16)] ; waves 0..3 take one M-subtile each
  if (wave < 4) {
    v8f acc = {};
#pragma unroll
    for (int kc = 0; kc < 4; ++kc)
      acc = wmma32(load_a_frag(&sAct[wave * 16][0], 128, kc * 32, l16, half_), v3f[kc], acc);
#pragma unroll
    for (int v = 0; v < 8; ++v)
      sH[wave * 16 + v + 8 * half_][l16] = acc[v] + sB[384 + l16];
  }
  __syncthreads();

  for (int idx = tid; idx < 64 * 14; idx += 256) {
    const int r = idx / 14, c = idx % 14;
    out[(m0 + r) * 14 + c] = sH[r][c];
  }
}

// ---------------------------------------------------------------------------
extern "C" void kernel_launch(void* const* d_in, const int* in_sizes, int n_in,
                              void* d_out, int out_size, void* d_ws, size_t ws_size,
                              hipStream_t stream) {
  const float* t    = (const float*)d_in[0];
  const float* ctx  = (const float*)d_in[1];
  const float* ce_w = (const float*)d_in[2];
  const float* ce_b = (const float*)d_in[3];
  const float* z0_w = (const float*)d_in[4];
  const float* z0_b = (const float*)d_in[5];
  const float* dw0 = (const float*)d_in[6];  const float* db0 = (const float*)d_in[7];
  const float* dw1 = (const float*)d_in[8];  const float* db1 = (const float*)d_in[9];
  const float* dw2 = (const float*)d_in[10]; const float* db2 = (const float*)d_in[11];
  const float* dw3 = (const float*)d_in[12]; const float* db3 = (const float*)d_in[13];
  const float* dg0 = (const float*)d_in[14]; const float* dbe0 = (const float*)d_in[15];
  const float* dg1 = (const float*)d_in[16]; const float* dbe1 = (const float*)d_in[17];
  const float* dg2 = (const float*)d_in[18]; const float* dbe2 = (const float*)d_in[19];
  const float* ew0 = (const float*)d_in[20]; const float* eb0 = (const float*)d_in[21];
  const float* ew1 = (const float*)d_in[22]; const float* eb1 = (const float*)d_in[23];
  const float* ew2 = (const float*)d_in[24]; const float* eb2 = (const float*)d_in[25];
  const float* ew3 = (const float*)d_in[26]; const float* eb3 = (const float*)d_in[27];
  const float* eg0 = (const float*)d_in[28]; const float* ebe0 = (const float*)d_in[29];
  const float* eg1 = (const float*)d_in[30]; const float* ebe1 = (const float*)d_in[31];
  const float* eg2 = (const float*)d_in[32]; const float* ebe2 = (const float*)d_in[33];

  _Float16* ztraj = (_Float16*)d_ws;  // 2048*256*64 f16 = 64 MiB
  float* out = (float*)d_out;

  rocket_scan_kernel<<<128, 256, 0, stream>>>(
      t, ctx, ce_w, ce_b, z0_w, z0_b,
      dw0, db0, dw1, db1, dw2, db2, dw3, db3,
      dg0, dbe0, dg1, dbe1, dg2, dbe2, ztraj);

  rocket_dec_kernel<<<8192, 256, 0, stream>>>(
      ztraj, ew0, eb0, ew1, eb1, ew2, eb2, ew3, eb3,
      eg0, ebe0, eg1, ebe1, eg2, ebe2, out);
}
